// GlobalLocalAwareEncoder_48361331753463
// MI455X (gfx1250) — matrix operand order repaired and checked by reading, not compiled, not run on hardware
//
#include <hip/hip_runtime.h>
#include <hip/hip_bf16.h>

// ---------------------------------------------------------------------------
// LongNet-style encoder for MI455X (gfx1250, wave32, WMMA + TDM).
// GEMMs/attention use v_wmma_f32_16x16x32_bf16 (fp32 accum); GEMM B panels
// are staged into LDS by the Tensor Data Mover, double-buffered.
// ---------------------------------------------------------------------------

typedef __bf16 bf16_t;
typedef __attribute__((ext_vector_type(16))) __bf16 v16bf;
typedef __attribute__((ext_vector_type(8)))  __bf16 v8bf;
typedef __attribute__((ext_vector_type(8)))  float  v8f;
typedef __attribute__((ext_vector_type(4)))  unsigned int u32x4;
typedef __attribute__((ext_vector_type(4)))  int i32x4;
typedef __attribute__((ext_vector_type(8)))  int i32x8;

static constexpr int CB   = 16;     // batch
static constexpr int CNC  = 4095;   // context count
static constexpr int CS   = 4096;   // sequence (1 + NC)
static constexpr int CIN  = 512;    // input dim
static constexpr int CD   = 256;    // model dim
static constexpr int CH   = 8;      // heads
static constexpr int CHD  = 32;     // head dim
static constexpr int CFFN = 1024;   // ffn dim

// ---------------------------------------------------------------------------
// small elementwise kernels
// ---------------------------------------------------------------------------
__global__ void f32_to_bf16_kernel(const float* __restrict__ src,
                                   bf16_t* __restrict__ dst, size_t n) {
  size_t i = (size_t)blockIdx.x * blockDim.x + threadIdx.x;
  if (i < n) dst[i] = (bf16_t)src[i];
}

// Convert + transpose: dst[n*K + k] = (bf16)src[k*N + n].
// Weights become [N][K] with K contiguous -> TDM tiles and WMMA B fragments
// are fully contiguous (ds_load_b128 instead of scalar u16 gathers).
__global__ void wt_transpose_kernel(const float* __restrict__ src,
                                    bf16_t* __restrict__ dst, int K, int N) {
  size_t i = (size_t)blockIdx.x * blockDim.x + threadIdx.x;
  if (i >= (size_t)K * N) return;
  int k = (int)(i / N), n = (int)(i % N);
  dst[(size_t)n * K + k] = (bf16_t)src[i];
}

__global__ void zero_f32_kernel(float* __restrict__ dst, size_t n) {
  size_t i = (size_t)blockIdx.x * blockDim.x + threadIdx.x;
  if (i < n) dst[i] = 0.0f;
}

// Build X_bf16[B*S, IN] = concat(target_feat, context_feats) in bf16.
__global__ void pack_x_kernel(const float* __restrict__ tf,
                              const float* __restrict__ cf,
                              bf16_t* __restrict__ Xb) {
  size_t i = (size_t)blockIdx.x * blockDim.x + threadIdx.x;  // over B*S*IN
  size_t row = i / CIN;
  int d = (int)(i % CIN);
  int b = (int)(row / CS), s = (int)(row % CS);
  float v = (s == 0) ? tf[(size_t)b * CIN + d]
                     : cf[((size_t)b * CNC + (s - 1)) * CIN + d];
  Xb[i] = (bf16_t)v;
}

// x += 0.3 * twoD_sincos(pos, 256)
__global__ void posenc_kernel(float* __restrict__ x,
                              const float* __restrict__ tp,
                              const float* __restrict__ cp) {
  size_t i = (size_t)blockIdx.x * blockDim.x + threadIdx.x;  // over B*S*D
  size_t row = i / CD;
  int d = (int)(i % CD);
  int b = (int)(row / CS), s = (int)(row % CS);
  const float* pp = (s == 0) ? (tp + (size_t)b * 2)
                             : (cp + ((size_t)b * CNC + (s - 1)) * 2);
  float pos = (d < 128) ? pp[0] : pp[1];
  int dd = d & 127;
  int j = dd & 63;
  float omega = __powf(10000.0f, -(float)j * (1.0f / 64.0f));
  float ang = pos * omega;
  float pe = (dd < 64) ? __sinf(ang) : __cosf(ang);
  x[i] += 0.3f * pe;
}

// ---------------------------------------------------------------------------
// LayerNorm: one wave per 256-float row (8 rows / 256-thread block).
// ---------------------------------------------------------------------------
__global__ __launch_bounds__(256)
void layernorm_bf16_kernel(const float* __restrict__ x,
                           const float* __restrict__ gamma,
                           const float* __restrict__ beta,
                           bf16_t* __restrict__ out) {
  int lane = threadIdx.x & 31;
  int wv   = threadIdx.x >> 5;
  size_t row = (size_t)blockIdx.x * 8 + wv;
  const float* xp = x + row * CD;
  float v[8];
  float s = 0.0f;
#pragma unroll
  for (int j = 0; j < 8; ++j) { v[j] = xp[lane + j * 32]; s += v[j]; }
#pragma unroll
  for (int m = 1; m < 32; m <<= 1) s += __shfl_xor(s, m, 32);
  float mu = s * (1.0f / CD);
  float q = 0.0f;
#pragma unroll
  for (int j = 0; j < 8; ++j) { float d = v[j] - mu; q += d * d; }
#pragma unroll
  for (int m = 1; m < 32; m <<= 1) q += __shfl_xor(q, m, 32);
  float rs = rsqrtf(q * (1.0f / CD) + 1e-5f);
  bf16_t* op = out + row * CD;
#pragma unroll
  for (int j = 0; j < 8; ++j) {
    int d = lane + j * 32;
    op[d] = (bf16_t)((v[j] - mu) * rs * gamma[d] + beta[d]);
  }
}

// Final LN: only rows b*S (token 0 of each batch) -> fp32 d_out[B, D].
__global__ __launch_bounds__(256)
void final_ln_kernel(const float* __restrict__ x,
                     const float* __restrict__ gamma,
                     const float* __restrict__ beta,
                     float* __restrict__ out) {
  int lane = threadIdx.x & 31;
  int wv   = threadIdx.x >> 5;
  int b = blockIdx.x * 8 + wv;
  if (b >= CB) return;
  const float* xp = x + (size_t)b * CS * CD;  // token 0 row
  float v[8];
  float s = 0.0f;
#pragma unroll
  for (int j = 0; j < 8; ++j) { v[j] = xp[lane + j * 32]; s += v[j]; }
#pragma unroll
  for (int m = 1; m < 32; m <<= 1) s += __shfl_xor(s, m, 32);
  float mu = s * (1.0f / CD);
  float q = 0.0f;
#pragma unroll
  for (int j = 0; j < 8; ++j) { float d = v[j] - mu; q += d * d; }
#pragma unroll
  for (int m = 1; m < 32; m <<= 1) q += __shfl_xor(q, m, 32);
  float rs = rsqrtf(q * (1.0f / CD) + 1e-5f);
  float* op = out + (size_t)b * CD;
#pragma unroll
  for (int j = 0; j < 8; ++j) {
    int d = lane + j * 32;
    op[d] = (v[j] - mu) * rs * gamma[d] + beta[d];
  }
}

// ---------------------------------------------------------------------------
// TDM: DMA a 64(n) x 32(k) bf16 panel of the transposed weight Wt[N][K]
// into LDS (row-major: sB[n*32 + k]). D# per CDNA5 ISA ch.8:
//   group0: count=1 | lds_addr | global_addr | type=2
//   group1: data_size=2B, tensor_dim0=K, tensor_dim1=N, tile=32x64,
//           tensor_dim0_stride=K
// ---------------------------------------------------------------------------
__device__ __forceinline__ void tdm_load_panel(const bf16_t* gptr,
                                               unsigned lds_off,
                                               int kdim, int ntotal) {
  unsigned long long ga = (unsigned long long)(size_t)gptr;
  u32x4 g0;
  g0[0] = 1u;                                              // count=1, user D#
  g0[1] = lds_off;                                         // lds_addr
  g0[2] = (unsigned)(ga & 0xFFFFFFFFull);                  // global_addr lo
  g0[3] = (unsigned)((ga >> 32) & 0x01FFFFFFull) | (2u << 30);  // hi | type=2
  i32x8 g1;
  g1[0] = (int)(1u << 16);                                 // data_size = 2B
  g1[1] = (int)(((unsigned)kdim & 0xFFFFu) << 16);         // tensor_dim0 lo16
  g1[2] = (int)(((unsigned)kdim >> 16) |
                (((unsigned)ntotal & 0xFFFFu) << 16));     // dim0 hi | dim1 lo
  g1[3] = (int)((((unsigned)ntotal >> 16) & 0xFFFFu) |
                (32u << 16));                              // dim1 hi | tile0=32
  g1[4] = 64;                                              // tile_dim1 = 64
  g1[5] = kdim;                                            // dim0_stride lo32
  g1[6] = 0;
  g1[7] = 0;
  i32x4 z4 = {0, 0, 0, 0};
#if __clang_major__ >= 23
  i32x8 z8 = {0, 0, 0, 0, 0, 0, 0, 0};
  __builtin_amdgcn_tensor_load_to_lds(g0, g1, z4, z4, z8, 0);
#else
  __builtin_amdgcn_tensor_load_to_lds(g0, g1, z4, z4, 0);
#endif
}

// ---------------------------------------------------------------------------
// bf16 WMMA GEMM:  C[M,N] = A[M,K] @ Wt[N,K]^T (+bias)(+residual)(gelu?)
// Block: 256 threads = 8 waves. Block tile 256x64, wave tile 32x64
// (2 A fragments x 4 B fragments -> 8 WMMAs per k-step; each B fragment
// feeds 2 WMMAs, each A fragment 4).
// B panel double-buffered in LDS via TDM (issued by wave 0, pipelined one
// panel ahead, published with s_wait_tensorcnt + barrier).
// M % 256 == 0, N % 64 == 0, K % 32 == 0 (true for all uses here).
// ---------------------------------------------------------------------------
__global__ __launch_bounds__(256)
void gemm_bf16_wmma(const bf16_t* __restrict__ A, int lda,
                    const bf16_t* __restrict__ Wt,     // [N][K], K contiguous
                    const float* __restrict__ bias,
                    const float* __restrict__ residual, int ldr,
                    float* __restrict__ Cf, bf16_t* __restrict__ Cb, int ldc,
                    int K, int N, int do_gelu) {
  __shared__ bf16_t sB[2][64 * 32];   // two 4KB panels
  const int lane = threadIdx.x & 31;
  const int wv   = threadIdx.x >> 5;
  const int grp  = lane >> 4;         // half-wave group
  const int lr   = lane & 15;
  const size_t m0 = (size_t)blockIdx.y * 256 + (size_t)wv * 32;
  const int n0 = blockIdx.x * 64;
  const int nsteps = K >> 5;

  v8f acc[2][4];
  const v8f vz = {0.f, 0.f, 0.f, 0.f, 0.f, 0.f, 0.f, 0.f};
#pragma unroll
  for (int rr = 0; rr < 2; ++rr)
#pragma unroll
    for (int ct = 0; ct < 4; ++ct) acc[rr][ct] = vz;

  const unsigned lds_base = (unsigned)(size_t)(&sB[0][0]);

  if (wv == 0)  // prologue: DMA panel 0
    tdm_load_panel(Wt + (size_t)n0 * K, lds_base, K, N);

  for (int step = 0; step < nsteps; ++step) {
    if (wv == 0) {
      if (step + 1 < nsteps) {
        tdm_load_panel(Wt + (size_t)n0 * K + (step + 1) * 32,
                       lds_base + (unsigned)(((step + 1) & 1) * 64 * 32 * 2),
                       K, N);
        __builtin_amdgcn_s_wait_tensorcnt((short)1);  // panel `step` done
      } else {
        __builtin_amdgcn_s_wait_tensorcnt((short)0);
      }
    }
    __syncthreads();

    const int k0 = step * 32;
    // A fragments (16 contiguous K per lane): two 16B global loads each
    v16bf af[2];
#pragma unroll
    for (int rr = 0; rr < 2; ++rr) {
      const bf16_t* ap =
          A + (m0 + rr * 16 + lr) * (size_t)lda + k0 + grp * 16;
      v8bf alo = *(const v8bf*)ap;
      v8bf ahi = *(const v8bf*)(ap + 8);
      af[rr] = __builtin_shufflevector(alo, ahi, 0, 1, 2, 3, 4, 5, 6, 7,
                                       8, 9, 10, 11, 12, 13, 14, 15);
    }

    const bf16_t* pb = &sB[step & 1][0];
#pragma unroll
    for (int ct = 0; ct < 4; ++ct) {
      // B fragment: 16 contiguous K per lane -> 2x ds_load_b128
      int col = ct * 16 + lr;
      v16bf bfv = *(const v16bf*)(pb + (size_t)col * 32 + grp * 16);
#pragma unroll
      for (int rr = 0; rr < 2; ++rr)
        acc[rr][ct] = __builtin_amdgcn_wmma_f32_16x16x32_bf16(
            false, af[rr], false, bfv, (short)0, acc[rr][ct], false, false);
    }
    __syncthreads();
  }

  // epilogue (C layout: row = grp*8 + i, col = lr)
#pragma unroll
  for (int rr = 0; rr < 2; ++rr) {
#pragma unroll
    for (int ct = 0; ct < 4; ++ct) {
      int col = n0 + ct * 16 + lr;
      float bv = bias ? bias[col] : 0.0f;
#pragma unroll
      for (int i = 0; i < 8; ++i) {
        size_t row = m0 + rr * 16 + grp * 8 + i;
        float v = acc[rr][ct][i] + bv;
        if (residual) v += residual[row * (size_t)ldr + col];
        if (do_gelu) {
          float u = v;
          v = 0.5f * u *
              (1.0f + tanhf(0.7978845608f * (u + 0.044715f * u * u * u)));
        }
        if (Cf) Cf[row * (size_t)ldc + col] = v;
        if (Cb) Cb[row * (size_t)ldc + col] = (bf16_t)v;
      }
    }
  }
}

// ---------------------------------------------------------------------------
// Dilated attention branch. One wave per (batch, segment, head).
// m = w/r = 64 tokens for every branch -> 4x4 grid of 16x16 WMMA score
// tiles (K=32=HD), in-register softmax via half-group shuffles, P staged
// through LDS in bf16, then P@V as 4x2x2 WMMAs. Adds o/3 into acc.
// ---------------------------------------------------------------------------
__global__ __launch_bounds__(32)
void attn_branch_kernel(const bf16_t* __restrict__ Q,
                        const bf16_t* __restrict__ K,
                        const bf16_t* __restrict__ V,
                        float* __restrict__ acc, int w, int r) {
  __shared__ bf16_t sP[64 * 64];
  const int lane = threadIdx.x;
  const int grp = lane >> 4, lr = lane & 15;
  const int seg = blockIdx.x, h = blockIdx.y, b = blockIdx.z;
  const int off = h % r;  // dilation offset for this head
  const size_t base = ((size_t)b * CS + (size_t)seg * w) * CD + (size_t)h * CHD;

  const v8f vz = {0.f, 0.f, 0.f, 0.f, 0.f, 0.f, 0.f, 0.f};
  v8f sc[4][4];
#pragma unroll
  for (int rt = 0; rt < 4; ++rt)
#pragma unroll
    for (int ct = 0; ct < 4; ++ct) sc[rt][ct] = vz;

  // Q fragments (A layout): two 16B loads each
  v16bf qf[4];
#pragma unroll
  for (int rt = 0; rt < 4; ++rt) {
    int j = rt * 16 + lr;
    const bf16_t* qp = Q + base + (size_t)(off + r * j) * CD + grp * 16;
    v8bf qlo = *(const v8bf*)qp;
    v8bf qhi = *(const v8bf*)(qp + 8);
    qf[rt] = __builtin_shufflevector(qlo, qhi, 0, 1, 2, 3, 4, 5, 6, 7,
                                     8, 9, 10, 11, 12, 13, 14, 15);
  }

  // S = Q @ K^T : B fragment = 32B contiguous (feature dim) per lane
#pragma unroll
  for (int ct = 0; ct < 4; ++ct) {
    int j = ct * 16 + lr;
    const bf16_t* kp = K + base + (size_t)(off + r * j) * CD + grp * 16;
    v16bf kf = *(const v16bf*)kp;
#pragma unroll
    for (int rt = 0; rt < 4; ++rt)
      sc[rt][ct] = __builtin_amdgcn_wmma_f32_16x16x32_bf16(
          false, qf[rt], false, kf, (short)0, sc[rt][ct], false, false);
  }

  // softmax over 64 columns; rows live in fixed 16-lane half-groups
  const float scale = 0.17677669529663687f;  // 1/sqrt(32)
#pragma unroll
  for (int rt = 0; rt < 4; ++rt) {
#pragma unroll
    for (int i = 0; i < 8; ++i) {
      float a0 = sc[rt][0][i] * scale, a1 = sc[rt][1][i] * scale;
      float a2 = sc[rt][2][i] * scale, a3 = sc[rt][3][i] * scale;
      float mx = fmaxf(fmaxf(a0, a1), fmaxf(a2, a3));
#pragma unroll
      for (int m = 1; m <= 8; m <<= 1) mx = fmaxf(mx, __shfl_xor(mx, m, 32));
      float e0 = __expf(a0 - mx), e1 = __expf(a1 - mx);
      float e2 = __expf(a2 - mx), e3 = __expf(a3 - mx);
      float sm = e0 + e1 + e2 + e3;
#pragma unroll
      for (int m = 1; m <= 8; m <<= 1) sm += __shfl_xor(sm, m, 32);
      float inv = 1.0f / sm;
      int row = rt * 16 + grp * 8 + i;
      sP[row * 64 + 0 * 16 + lr] = (bf16_t)(e0 * inv);
      sP[row * 64 + 1 * 16 + lr] = (bf16_t)(e1 * inv);
      sP[row * 64 + 2 * 16 + lr] = (bf16_t)(e2 * inv);
      sP[row * 64 + 3 * 16 + lr] = (bf16_t)(e3 * inv);
    }
  }
  __syncthreads();

  // O = P @ V : K=64 -> 2 k-tiles; HD=32 -> 2 col-tiles
  v8f o[4][2];
#pragma unroll
  for (int rt = 0; rt < 4; ++rt) { o[rt][0] = vz; o[rt][1] = vz; }

#pragma unroll
  for (int kt = 0; kt < 2; ++kt) {
    v16bf pf[4];
#pragma unroll
    for (int rt = 0; rt < 4; ++rt) {
      const bf16_t* pp = sP + (rt * 16 + lr) * 64 + kt * 32 + grp * 8;
      v8bf plo = *(const v8bf*)pp;
      v8bf phi = *(const v8bf*)(pp + 16);
      pf[rt] = __builtin_shufflevector(plo, phi, 0, 1, 2, 3, 4, 5, 6, 7,
                                       8, 9, 10, 11, 12, 13, 14, 15);
    }
#pragma unroll
    for (int c2 = 0; c2 < 2; ++c2) {
      int c = c2 * 16 + lr;
      v16bf vf;
#pragma unroll
      for (int jj = 0; jj < 16; ++jj) {
        int krow = kt * 32 + grp * 8 + (jj & 7) + (jj >> 3) * 16;
        vf[jj] = V[base + (size_t)(off + r * krow) * CD + c];
      }
#pragma unroll
      for (int rt = 0; rt < 4; ++rt)
        o[rt][c2] = __builtin_amdgcn_wmma_f32_16x16x32_bf16(
            false, pf[rt], false, vf, (short)0, o[rt][c2], false, false);
    }
  }

  // accumulate (branch average 1/3 folded in); within one branch each
  // (b, s, h, hd) is written by exactly one wave -> plain RMW is safe.
#pragma unroll
  for (int rt = 0; rt < 4; ++rt)
#pragma unroll
    for (int c2 = 0; c2 < 2; ++c2)
#pragma unroll
      for (int i = 0; i < 8; ++i) {
        int m = rt * 16 + grp * 8 + i;
        size_t outp = base + (size_t)(off + r * m) * CD + c2 * 16 + lr;
        acc[outp] += o[rt][c2][i] * (1.0f / 3.0f);
      }
}

// ---------------------------------------------------------------------------
// host orchestration
// ---------------------------------------------------------------------------
extern "C" void kernel_launch(void* const* d_in, const int* in_sizes, int n_in,
                              void* d_out, int out_size, void* d_ws, size_t ws_size,
                              hipStream_t stream) {
  (void)in_sizes; (void)n_in; (void)out_size; (void)ws_size;

  const float* target_feat   = (const float*)d_in[0];
  const float* context_feats = (const float*)d_in[1];
  const float* target_pos    = (const float*)d_in[2];
  const float* context_pos   = (const float*)d_in[3];
  const float* W_proj        = (const float*)d_in[4];
  const float* b_proj        = (const float*)d_in[5];
  const float* Wq            = (const float*)d_in[6];
  const float* Wk            = (const float*)d_in[7];
  const float* Wv            = (const float*)d_in[8];
  const float* Wo            = (const float*)d_in[9];
  const float* ln1_s         = (const float*)d_in[10];
  const float* ln1_b         = (const float*)d_in[11];
  const float* ln2_s         = (const float*)d_in[12];
  const float* ln2_b         = (const float*)d_in[13];
  const float* W1            = (const float*)d_in[14];
  const float* b1            = (const float*)d_in[15];
  const float* W2            = (const float*)d_in[16];
  const float* b2            = (const float*)d_in[17];
  const float* lnf_s         = (const float*)d_in[18];
  const float* lnf_b         = (const float*)d_in[19];
  float* out                 = (float*)d_out;

  const size_t ROWS = (size_t)CB * CS;  // 65536

  // workspace carving (deterministic)
  char* p = (char*)d_ws;
  auto take = [&](size_t bytes) -> char* {
    char* r = p;
    p += (bytes + 255) & ~(size_t)255;
    return r;
  };
  bf16_t* Wp_t = (bf16_t*)take((size_t)CIN * CD * 2);    // [D][IN]
  bf16_t* Wq_t = (bf16_t*)take((size_t)2 * CD * CD * 2); // [D][D] per layer
  bf16_t* Wk_t = (bf16_t*)take((size_t)2 * CD * CD * 2);
  bf16_t* Wv_t = (bf16_t*)take((size_t)2 * CD * CD * 2);
  bf16_t* Wo_t = (bf16_t*)take((size_t)2 * CD * CD * 2);
  bf16_t* W1_t = (bf16_t*)take((size_t)2 * CD * CFFN * 2);   // [FFN][D]
  bf16_t* W2_t = (bf16_t*)take((size_t)2 * CFFN * CD * 2);   // [D][FFN]
  bf16_t* Xb   = (bf16_t*)take(ROWS * CIN * 2);
  float*  x    = (float*) take(ROWS * CD * 4);
  bf16_t* lnb  = (bf16_t*)take(ROWS * CD * 2);
  bf16_t* Qb   = (bf16_t*)take(ROWS * CD * 2);
  bf16_t* Kb   = (bf16_t*)take(ROWS * CD * 2);
  bf16_t* Vb   = (bf16_t*)take(ROWS * CD * 2);
  float*  accf = (float*) take(ROWS * CD * 4);
  bf16_t* accb = (bf16_t*)take(ROWS * CD * 2);
  bf16_t* g    = (bf16_t*)take(ROWS * CFFN * 2);

  // weights -> bf16, transposed to [N][K]
  auto cvtT = [&](const float* s, bf16_t* d, int K, int N) {
    size_t n = (size_t)K * N;
    wt_transpose_kernel<<<(unsigned)((n + 255) / 256), 256, 0, stream>>>(s, d, K, N);
  };
  cvtT(W_proj, Wp_t, CIN, CD);
  for (int l = 0; l < 2; ++l) {
    cvtT(Wq + (size_t)l * CD * CD, Wq_t + (size_t)l * CD * CD, CD, CD);
    cvtT(Wk + (size_t)l * CD * CD, Wk_t + (size_t)l * CD * CD, CD, CD);
    cvtT(Wv + (size_t)l * CD * CD, Wv_t + (size_t)l * CD * CD, CD, CD);
    cvtT(Wo + (size_t)l * CD * CD, Wo_t + (size_t)l * CD * CD, CD, CD);
    cvtT(W1 + (size_t)l * CD * CFFN, W1_t + (size_t)l * CD * CFFN, CD, CFFN);
    cvtT(W2 + (size_t)l * CFFN * CD, W2_t + (size_t)l * CFFN * CD, CFFN, CD);
  }

  // pack inputs as bf16
  pack_x_kernel<<<(unsigned)(ROWS * CIN / 256), 256, 0, stream>>>(
      target_feat, context_feats, Xb);

  // x = X @ Wp + b_proj
  gemm_bf16_wmma<<<dim3(CD / 64, (unsigned)(ROWS / 256)), 256, 0, stream>>>(
      Xb, CIN, Wp_t, b_proj, nullptr, 0, x, nullptr, CD, CIN, CD, 0);

  // x += 0.3 * 2D sincos positional encoding
  posenc_kernel<<<(unsigned)(ROWS * CD / 256), 256, 0, stream>>>(
      x, target_pos, context_pos);

  for (int l = 0; l < 2; ++l) {
    const bf16_t* Wq_l = Wq_t + (size_t)l * CD * CD;
    const bf16_t* Wk_l = Wk_t + (size_t)l * CD * CD;
    const bf16_t* Wv_l = Wv_t + (size_t)l * CD * CD;
    const bf16_t* Wo_l = Wo_t + (size_t)l * CD * CD;
    const bf16_t* W1_l = W1_t + (size_t)l * CD * CFFN;
    const bf16_t* W2_l = W2_t + (size_t)l * CFFN * CD;

    // ln1
    layernorm_bf16_kernel<<<(unsigned)(ROWS / 8), 256, 0, stream>>>(
        x, ln1_s + l * CD, ln1_b + l * CD, lnb);

    // Q, K, V (bf16 outputs feed attention WMMAs)
    gemm_bf16_wmma<<<dim3(CD / 64, (unsigned)(ROWS / 256)), 256, 0, stream>>>(
        lnb, CD, Wq_l, nullptr, nullptr, 0, nullptr, Qb, CD, CD, CD, 0);
    gemm_bf16_wmma<<<dim3(CD / 64, (unsigned)(ROWS / 256)), 256, 0, stream>>>(
        lnb, CD, Wk_l, nullptr, nullptr, 0, nullptr, Kb, CD, CD, CD, 0);
    gemm_bf16_wmma<<<dim3(CD / 64, (unsigned)(ROWS / 256)), 256, 0, stream>>>(
        lnb, CD, Wv_l, nullptr, nullptr, 0, nullptr, Vb, CD, CD, CD, 0);

    // dilated attention: acc = mean of 3 branches
    zero_f32_kernel<<<(unsigned)(ROWS * CD / 256), 256, 0, stream>>>(
        accf, ROWS * CD);
    attn_branch_kernel<<<dim3(CS / 64, CH, CB), 32, 0, stream>>>(
        Qb, Kb, Vb, accf, 64, 1);
    attn_branch_kernel<<<dim3(CS / 128, CH, CB), 32, 0, stream>>>(
        Qb, Kb, Vb, accf, 128, 2);
    attn_branch_kernel<<<dim3(CS / 256, CH, CB), 32, 0, stream>>>(
        Qb, Kb, Vb, accf, 256, 4);
    f32_to_bf16_kernel<<<(unsigned)(ROWS * CD / 256), 256, 0, stream>>>(
        accf, accb, ROWS * CD);

    // x = x + attn @ Wo
    gemm_bf16_wmma<<<dim3(CD / 64, (unsigned)(ROWS / 256)), 256, 0, stream>>>(
        accb, CD, Wo_l, nullptr, x, CD, x, nullptr, CD, CD, CD, 0);

    // ln2
    layernorm_bf16_kernel<<<(unsigned)(ROWS / 8), 256, 0, stream>>>(
        x, ln2_s + l * CD, ln2_b + l * CD, lnb);

    // g = gelu(ln2 @ W1 + b1)
    gemm_bf16_wmma<<<dim3(CFFN / 64, (unsigned)(ROWS / 256)), 256, 0, stream>>>(
        lnb, CD, W1_l, b1 + l * CFFN, nullptr, 0, nullptr, g, CFFN, CD, CFFN, 1);

    // x = x + g @ W2 + b2
    gemm_bf16_wmma<<<dim3(CD / 64, (unsigned)(ROWS / 256)), 256, 0, stream>>>(
        g, CFFN, W2_l, b2 + l * CD, x, CD, x, nullptr, CD, CFFN, CD, 0);
  }

  // final LN on token 0 of each batch -> out[B, D]
  final_ln_kernel<<<2, 256, 0, stream>>>(x, lnf_s, lnf_b, out);
}
